// TransformerBlock_29472065585405
// MI455X (gfx1250) — matrix operand order repaired and checked by reading, not compiled, not run on hardware
//
#include <hip/hip_runtime.h>
#include <hip/hip_bf16.h>

// ---------------------------------------------------------------------------
// Types for WMMA fragments (gfx1250, wave32)
// ---------------------------------------------------------------------------
typedef _Float16 half_t;
typedef __attribute__((ext_vector_type(16))) half_t v16h;
typedef __attribute__((ext_vector_type(8)))  half_t v8h;
typedef __attribute__((ext_vector_type(8)))  float  v8f;

union Frag16 {
    v16h  v;
    v8h   h[2];
    half_t e[16];
};

static __device__ __forceinline__ v8f vzero8() {
    v8f z = {0.f, 0.f, 0.f, 0.f, 0.f, 0.f, 0.f, 0.f};
    return z;
}

static __device__ __forceinline__ v8f wmma_f16(v16h a, v16h b, v8f c) {
    // D = A(16x32 f16) * B(32x16 f16) + C(16x16 f32)
    return __builtin_amdgcn_wmma_f32_16x16x32_f16(
        /*neg_a=*/false, a, /*neg_b=*/false, b,
        /*c_mod=*/(short)0, c, /*reuse_a=*/false, /*reuse_b=*/false);
}

// gfx1250 async copy: per-lane LDS[ldsAddr] = MEM[gaddr] (16 bytes), ASYNCcnt
static __device__ __forceinline__ void async_g2lds_b128(unsigned ldsAddr, const void* gaddr) {
    asm volatile("global_load_async_to_lds_b128 %0, %1, off"
                 :: "v"(ldsAddr), "v"(gaddr) : "memory");
}
static __device__ __forceinline__ void wait_asynccnt0() {
    asm volatile("s_wait_asynccnt 0x0" ::: "memory");
}
// low 32 bits of a generic LDS pointer == LDS byte offset (aperture addressing)
static __device__ __forceinline__ unsigned lds_addr32(const void* p) {
    return (unsigned)(unsigned long long)p;
}

// ---------------------------------------------------------------------------
// Problem constants
// ---------------------------------------------------------------------------
#define BB    2
#define TT    2048
#define CC    1024
#define HH    16
#define DD    64
#define ROWS  (BB * TT)          // 4096

// ---------------------------------------------------------------------------
// f32 -> f16 conversion (weights)
// ---------------------------------------------------------------------------
__global__ void cvt_f32_to_f16(const float* __restrict__ s, half_t* __restrict__ d, int n) {
    int i = blockIdx.x * 256 + threadIdx.x;
    if (i < n) d[i] = (half_t)s[i];
}

// ---------------------------------------------------------------------------
// LayerNorm (per-row) -> f16 output
// ---------------------------------------------------------------------------
__launch_bounds__(256)
__global__ void layernorm_f16(const float* __restrict__ x, const float* __restrict__ g,
                              const float* __restrict__ be, half_t* __restrict__ out) {
    __shared__ float red[256];
    __shared__ float red2[256];
    const int row = blockIdx.x;
    const float* xr = x + (size_t)row * CC;
    float s = 0.f, s2 = 0.f;
    for (int c = threadIdx.x; c < CC; c += 256) {
        float v = xr[c];
        s += v; s2 += v * v;
    }
    red[threadIdx.x] = s; red2[threadIdx.x] = s2;
    __syncthreads();
    for (int off = 128; off > 0; off >>= 1) {
        if (threadIdx.x < off) {
            red[threadIdx.x]  += red[threadIdx.x + off];
            red2[threadIdx.x] += red2[threadIdx.x + off];
        }
        __syncthreads();
    }
    const float mean = red[0] * (1.0f / CC);
    const float var  = red2[0] * (1.0f / CC) - mean * mean;
    const float rstd = rsqrtf(var + 1e-5f);
    half_t* orow = out + (size_t)row * CC;
    for (int c = threadIdx.x; c < CC; c += 256)
        orow[c] = (half_t)((xr[c] - mean) * rstd * g[c] + be[c]);
}

// ---------------------------------------------------------------------------
// WMMA GEMM: out = act(A[MxK]f16 * B[KxN]f16 + bias (+ residual))
// Block tile 128x128, 8 waves (4 row-groups x 2 col-groups); each wave owns a
// 32x64 tile = acc[2][4] (8 WMMAs / K-step, B fragment reused twice).
// B tile (32 x 128) staged transposed in LDS (stride 40 halves -> all
// fragment reads are 16B-aligned ds_load_b128).
// Epilogue specialization via template: no runtime branches.
// ---------------------------------------------------------------------------
template <bool HasRes, bool Gelu, bool OutF32>
__launch_bounds__(256)
__global__ void gemm_wmma(const half_t* __restrict__ A, const half_t* __restrict__ Bm,
                          const float* __restrict__ bias, const float* __restrict__ residual,
                          void* __restrict__ outPtr, int M, int N, int K) {
    __shared__ half_t ldsB[128 * 40];

    const int tid  = threadIdx.x;
    const int wave = tid >> 5;
    const int lane = tid & 31;
    const int hf   = lane >> 4;     // half-wave: 0 or 1
    const int ln   = lane & 15;
    const int wr   = wave & 3;      // wave row-group 0..3 (32 rows each)
    const int wc   = wave >> 2;     // wave col-group 0..1 (64 cols each)
    const int M0   = blockIdx.y * 128;
    const int N0   = blockIdx.x * 128;
    const int rowA0 = M0 + wr * 32 + ln;
    const int rowA1 = rowA0 + 16;

    v8f acc[2][4] = {{vzero8(), vzero8(), vzero8(), vzero8()},
                     {vzero8(), vzero8(), vzero8(), vzero8()}};

    const int kkRow = tid >> 3;     // 0..31 : k-row of B tile this thread loads
    const int nb    = tid & 7;      // 0..7  : 16-col group

    for (int k0 = 0; k0 < K; k0 += 32) {
        // --- cooperative load of B[ k0..k0+31 ][ N0..N0+127 ] ---
        const half_t* bsrc = Bm + (size_t)(k0 + kkRow) * N + N0 + nb * 16;
        v8h b0 = *(const v8h*)bsrc;
        v8h b1 = *(const v8h*)(bsrc + 8);
        // prefetch next K-step's B tile row (speculative, gfx1250 global_prefetch)
        __builtin_prefetch(bsrc + (size_t)32 * N, 0, 1);
        __syncthreads();   // previous iteration's LDS reads complete
        #pragma unroll
        for (int j = 0; j < 8; ++j) ldsB[(nb * 16 + j) * 40 + kkRow]     = b0[j];
        #pragma unroll
        for (int j = 0; j < 8; ++j) ldsB[(nb * 16 + 8 + j) * 40 + kkRow] = b1[j];
        __syncthreads();

        // --- two A fragments (rows rowA0, rowA1), k(e) = k0 + 8*hf + e (+8 for e>=8)
        Frag16 a0, a1;
        const half_t* asrc0 = A + (size_t)rowA0 * K + k0 + 8 * hf;
        const half_t* asrc1 = A + (size_t)rowA1 * K + k0 + 8 * hf;
        a0.h[0] = *(const v8h*)asrc0;
        a0.h[1] = *(const v8h*)(asrc0 + 16);
        a1.h[0] = *(const v8h*)asrc1;
        a1.h[1] = *(const v8h*)(asrc1 + 16);
        __builtin_prefetch(asrc0 + 32, 0, 1);

        // --- 4 B fragments from LDS, each reused for both A row-groups ---
        #pragma unroll
        for (int j = 0; j < 4; ++j) {
            Frag16 b;
            const half_t* bl = &ldsB[(wc * 64 + j * 16 + ln) * 40 + 8 * hf];
            b.h[0] = *(const v8h*)bl;
            b.h[1] = *(const v8h*)(bl + 16);
            acc[0][j] = wmma_f16(a0.v, b.v, acc[0][j]);
            acc[1][j] = wmma_f16(a1.v, b.v, acc[1][j]);
        }
    }

    // --- epilogue: bias [+ residual] [gelu], store f32 or f16 ---
    float*  outF = (float*)outPtr;
    half_t* outH = (half_t*)outPtr;
    #pragma unroll
    for (int i = 0; i < 2; ++i) {
        #pragma unroll
        for (int j = 0; j < 4; ++j) {
            const int col = N0 + wc * 64 + j * 16 + ln;
            const float bv = bias[col];
            #pragma unroll
            for (int r = 0; r < 8; ++r) {
                const int row = M0 + wr * 32 + i * 16 + r + 8 * hf;
                float vv = acc[i][j][r] + bv;
                if (HasRes) vv += residual[(size_t)row * N + col];
                if (Gelu)   vv = 0.5f * vv * (1.0f + erff(vv * 0.70710678118654752f));
                if (OutF32) outF[(size_t)row * N + col] = vv;
                else        outH[(size_t)row * N + col] = (half_t)vv;
            }
        }
    }
}

// ---------------------------------------------------------------------------
// Split fused QKV [B*T, 3C] (f16) into Q/K/V [B,H,T,D] (f16); fold 1/sqrt(d)
// into Q.
// ---------------------------------------------------------------------------
__global__ void qkv_split(const half_t* __restrict__ qkv, half_t* __restrict__ q,
                          half_t* __restrict__ k, half_t* __restrict__ v) {
    const int idx = blockIdx.x * 256 + threadIdx.x;   // over B*T*C = 2^22
    if (idx >= BB * TT * CC) return;
    const int d = idx & 63;
    const int h = (idx >> 6) & 15;
    const int t = (idx >> 10) & 2047;
    const int b = idx >> 21;
    const size_t src = (size_t)(b * TT + t) * (3 * CC);
    const size_t dst = ((size_t)(b * HH + h) * TT + t) * DD + d;
    const int c = h * DD + d;
    q[dst] = (half_t)((float)qkv[src + c] * 0.125f);   // 1/sqrt(64)
    k[dst] = qkv[src + CC + c];
    v[dst] = qkv[src + 2 * CC + c];
}

// ---------------------------------------------------------------------------
// Flash-style causal attention.
// Block = 8 waves covering 8 consecutive 16-row query tiles of one (b,h).
// Per 32-key block: K and V tiles (32x64 f16 = 4096 B each = 256 lanes x 16 B)
// staged into LDS with gfx1250 async copies (global_load_async_to_lds_b128,
// fenced by s_wait_asynccnt before the barrier). Then each wave:
//   S = Q K^T (4 WMMAs, K frags as ds_load_b128), causal mask, online
//   softmax (shfl_xor row reductions within 16-lane half-groups), P bounced
//   through wave-private LDS (C-layout -> A-layout), O += P V (4 WMMAs).
// Waves past their causal limit only participate in barriers.
// ---------------------------------------------------------------------------
__launch_bounds__(256)
__global__ void attn_flash(const half_t* __restrict__ q, const half_t* __restrict__ k,
                           const half_t* __restrict__ v, half_t* __restrict__ yOut) {
    __shared__ half_t ldsK[32 * 64];
    __shared__ half_t ldsV[32 * 64];
    __shared__ half_t ldsP[8][16 * 32];

    const int tid  = threadIdx.x;
    const int wave = tid >> 5;
    const int lane = tid & 31;
    const int hf   = lane >> 4;
    const int ln   = lane & 15;
    const int bh   = blockIdx.y;          // 0..31
    const int b    = bh >> 4;
    const int h    = bh & 15;
    const int qt   = blockIdx.x * 8 + wave;   // query tile 0..127
    const int q0   = qt * 16;

    const half_t* qbase = q + (size_t)bh * TT * DD;
    const half_t* kbase = k + (size_t)bh * TT * DD;
    const half_t* vbase = v + (size_t)bh * TT * DD;

    // Q A-fragments for d0 = 0 and 32 (K-dim split of D=64)
    Frag16 qf[2];
    #pragma unroll
    for (int d0i = 0; d0i < 2; ++d0i) {
        const half_t* src = qbase + (size_t)(q0 + ln) * DD + d0i * 32 + 8 * hf;
        qf[d0i].h[0] = *(const v8h*)src;
        qf[d0i].h[1] = *(const v8h*)(src + 16);
    }

    v8f o[4] = {vzero8(), vzero8(), vzero8(), vzero8()};
    float mrun[8], lrun[8];
    #pragma unroll
    for (int r = 0; r < 8; ++r) { mrun[r] = -1e30f; lrun[r] = 0.f; }

    half_t* myP = &ldsP[wave][0];
    // all key blocks needed by the LAST tile of this block
    const int nkb = (blockIdx.x * 128 + 128) >> 5;
    const int myLimit = q0 + 15;          // keys <= myLimit are relevant

    const int ldRow = tid >> 3;           // 0..31
    const int ldCol = (tid & 7) * 8;      // 0,8,...,56
    const unsigned ldsKAddr = lds_addr32(&ldsK[ldRow * 64 + ldCol]);
    const unsigned ldsVAddr = lds_addr32(&ldsV[ldRow * 64 + ldCol]);

    for (int kb = 0; kb < nkb; ++kb) {
        const int kBase = kb * 32;

        // --- async staging of K & V tiles [32 keys x 64 d] into LDS ---
        async_g2lds_b128(ldsKAddr, kbase + (size_t)(kBase + ldRow) * DD + ldCol);
        async_g2lds_b128(ldsVAddr, vbase + (size_t)(kBase + ldRow) * DD + ldCol);
        wait_asynccnt0();      // own wave's async copies landed in LDS
        __syncthreads();       // all waves' copies visible

        if (kBase <= myLimit) {
            v8f s0 = vzero8(), s1 = vzero8();

            // S tile 16x32 = two 16x16 fragments; K-loop over d (2x32)
            #pragma unroll
            for (int d0i = 0; d0i < 2; ++d0i) {
                Frag16 kf0, kf1;
                const half_t* src0 = &ldsK[(0 * 16 + ln) * 64 + d0i * 32 + 8 * hf];
                const half_t* src1 = &ldsK[(1 * 16 + ln) * 64 + d0i * 32 + 8 * hf];
                kf0.h[0] = *(const v8h*)src0;
                kf0.h[1] = *(const v8h*)(src0 + 16);
                kf1.h[0] = *(const v8h*)src1;
                kf1.h[1] = *(const v8h*)(src1 + 16);
                s0 = wmma_f16(qf[d0i].v, kf0.v, s0);
                s1 = wmma_f16(qf[d0i].v, kf1.v, s1);
            }

            // causal mask + online softmax update
            float p0[8], p1[8];
            #pragma unroll
            for (int r = 0; r < 8; ++r) {
                const int qrow = q0 + r + 8 * hf;
                float e0 = ((kBase + ln)      <= qrow) ? s0[r] : -1e30f;
                float e1 = ((kBase + 16 + ln) <= qrow) ? s1[r] : -1e30f;
                float t = fmaxf(e0, e1);
                t = fmaxf(t, __shfl_xor(t, 1, 32));
                t = fmaxf(t, __shfl_xor(t, 2, 32));
                t = fmaxf(t, __shfl_xor(t, 4, 32));
                t = fmaxf(t, __shfl_xor(t, 8, 32));
                const float mnew  = fmaxf(mrun[r], t);
                const float alpha = __expf(mrun[r] - mnew);
                const float pe0 = __expf(e0 - mnew);
                const float pe1 = __expf(e1 - mnew);
                float rs = pe0 + pe1;
                rs += __shfl_xor(rs, 1, 32);
                rs += __shfl_xor(rs, 2, 32);
                rs += __shfl_xor(rs, 4, 32);
                rs += __shfl_xor(rs, 8, 32);
                lrun[r] = lrun[r] * alpha + rs;
                mrun[r] = mnew;
                p0[r] = pe0; p1[r] = pe1;
                #pragma unroll
                for (int j = 0; j < 4; ++j) o[j][r] *= alpha;
            }

            // P: C-layout -> wave-private LDS (16x32 f16) -> A-layout fragment
            #pragma unroll
            for (int r = 0; r < 8; ++r) {
                const int row = r + 8 * hf;
                myP[row * 32 + ln]      = (half_t)p0[r];
                myP[row * 32 + 16 + ln] = (half_t)p1[r];
            }
            Frag16 pa;
            pa.h[0] = *(const v8h*)&myP[ln * 32 + 16 * hf];
            pa.h[1] = *(const v8h*)&myP[ln * 32 + 16 * hf + 8];

            // O += P * V  (V B-fragments from LDS: k(e)=16*hf+e, n=ln)
            #pragma unroll
            for (int j = 0; j < 4; ++j) {
                Frag16 vf;
                #pragma unroll
                for (int e = 0; e < 16; ++e)
                    vf.e[e] = ldsV[(16 * hf + e) * 64 + j * 16 + ln];
                o[j] = wmma_f16(pa.v, vf.v, o[j]);
            }
        }
        __syncthreads();   // before next iteration overwrites K/V tiles
    }

    // normalize by l and write f16 [B,T,C] at column h*64 + d
    float linv[8];
    #pragma unroll
    for (int r = 0; r < 8; ++r) linv[r] = __frcp_rn(lrun[r]);
    #pragma unroll
    for (int j = 0; j < 4; ++j) {
        #pragma unroll
        for (int r = 0; r < 8; ++r) {
            const int trow = q0 + r + 8 * hf;
            const float val = o[j][r] * linv[r];
            yOut[((size_t)(b * TT + trow)) * CC + h * DD + j * 16 + ln] = (half_t)val;
        }
    }
}

// ---------------------------------------------------------------------------
// Host-side orchestration
// ---------------------------------------------------------------------------
extern "C" void kernel_launch(void* const* d_in, const int* in_sizes, int n_in,
                              void* d_out, int out_size, void* d_ws, size_t ws_size,
                              hipStream_t stream) {
    (void)in_sizes; (void)n_in; (void)out_size; (void)ws_size;

    const float* x    = (const float*)d_in[0];
    const float* Wqkv = (const float*)d_in[1];
    const float* bqkv = (const float*)d_in[2];
    const float* Wo   = (const float*)d_in[3];
    const float* bo   = (const float*)d_in[4];
    const float* W1   = (const float*)d_in[5];
    const float* b1   = (const float*)d_in[6];
    const float* W2   = (const float*)d_in[7];
    const float* b2   = (const float*)d_in[8];
    const float* g1   = (const float*)d_in[9];
    const float* be1  = (const float*)d_in[10];
    const float* g2   = (const float*)d_in[11];
    const float* be2  = (const float*)d_in[12];
    float* out = (float*)d_out;

    // workspace layout (all sizes multiples of 256B)
    char* ws = (char*)d_ws;
    size_t off = 0;
    auto take = [&](size_t bytes) { char* p = ws + off; off += bytes; return p; };
    half_t* WqkvH = (half_t*)take((size_t)CC * 3 * CC * 2);
    half_t* WoH   = (half_t*)take((size_t)CC * CC * 2);
    half_t* W1H   = (half_t*)take((size_t)CC * 4 * CC * 2);
    half_t* W2H   = (half_t*)take((size_t)4 * CC * CC * 2);
    half_t* h1H   = (half_t*)take((size_t)ROWS * CC * 2);
    half_t* qkvH  = (half_t*)take((size_t)ROWS * 3 * CC * 2);
    half_t* qH    = (half_t*)take((size_t)ROWS * CC * 2);
    half_t* kH    = (half_t*)take((size_t)ROWS * CC * 2);
    half_t* vH    = (half_t*)take((size_t)ROWS * CC * 2);
    half_t* aYH   = (half_t*)take((size_t)ROWS * CC * 2);
    float*  yF    = (float*)take((size_t)ROWS * CC * 4);
    half_t* h2H   = (half_t*)take((size_t)ROWS * CC * 2);
    half_t* fH    = (half_t*)take((size_t)ROWS * 4 * CC * 2);

    // 1) weight conversions
    cvt_f32_to_f16<<<(CC * 3 * CC + 255) / 256, 256, 0, stream>>>(Wqkv, WqkvH, CC * 3 * CC);
    cvt_f32_to_f16<<<(CC * CC + 255) / 256, 256, 0, stream>>>(Wo, WoH, CC * CC);
    cvt_f32_to_f16<<<(CC * 4 * CC + 255) / 256, 256, 0, stream>>>(W1, W1H, CC * 4 * CC);
    cvt_f32_to_f16<<<(4 * CC * CC + 255) / 256, 256, 0, stream>>>(W2, W2H, 4 * CC * CC);

    // 2) LN1
    layernorm_f16<<<ROWS, 256, 0, stream>>>(x, g1, be1, h1H);

    // 3) QKV GEMM: [4096,1024] x [1024,3072] -> f16
    gemm_wmma<false, false, false><<<dim3(3 * CC / 128, ROWS / 128), 256, 0, stream>>>(
        h1H, WqkvH, bqkv, nullptr, qkvH, ROWS, 3 * CC, CC);

    // 4) split Q/K/V
    qkv_split<<<(BB * TT * CC + 255) / 256, 256, 0, stream>>>(qkvH, qH, kH, vH);

    // 5) causal flash attention -> f16 [B,T,C]
    attn_flash<<<dim3(TT / 16 / 8, BB * HH), 256, 0, stream>>>(qH, kH, vH, aYH);

    // 6) O-proj + residual(x) -> f32 y
    gemm_wmma<true, false, true><<<dim3(CC / 128, ROWS / 128), 256, 0, stream>>>(
        aYH, WoH, bo, x, yF, ROWS, CC, CC);

    // 7) LN2
    layernorm_f16<<<ROWS, 256, 0, stream>>>(yF, g2, be2, h2H);

    // 8) MLP up + exact GELU -> f16
    gemm_wmma<false, true, false><<<dim3(4 * CC / 128, ROWS / 128), 256, 0, stream>>>(
        h2H, W1H, b1, nullptr, fH, ROWS, 4 * CC, CC);

    // 9) MLP down + residual(y) -> f32 output
    gemm_wmma<true, false, true><<<dim3(CC / 128, ROWS / 128), 256, 0, stream>>>(
        fH, W2H, b2, yF, out, ROWS, CC, 4 * CC);
}
